// UpModConvBlock_80693845557900
// MI455X (gfx1250) — matrix operand order repaired and checked by reading, not compiled, not run on hardware
//
#include <hip/hip_runtime.h>
#include <hip/hip_bf16.h>

// ---------- WMMA vector types (gfx1250, wave32) ----------
typedef __attribute__((ext_vector_type(16))) __bf16 v16bf;
typedef __attribute__((ext_vector_type(8)))  float  v8f;

#define B_      8
#define IN_CH   512
#define OUT_CH  256
#define HW      64
#define HP      66    // padded spatial (1-px zero halo each side)
#define CT      129   // conv-transpose output spatial size
#define OUTHW   128

__device__ __forceinline__ unsigned short f32_to_bf16(float f) {
    unsigned int u = __float_as_uint(f);
    unsigned int r = u + 0x7fffu + ((u >> 16) & 1u);   // round-to-nearest-even
    return (unsigned short)(r >> 16);
}

// ---------------------------------------------------------------------------
// K0: zero-fill the padded channels-last input buffer (halo must be 0)
// ---------------------------------------------------------------------------
__global__ __launch_bounds__(256)
void zero_kernel(uint4* __restrict__ p, unsigned int n16) {
    unsigned int i = blockIdx.x * 256 + threadIdx.x;
    if (i < n16) p[i] = make_uint4(0u, 0u, 0u, 0u);
}

// ---------------------------------------------------------------------------
// K0b: transpose (8,512,64,64) f32 -> channels-last bf16 (8,66,66,512), halo kept 0.
// LDS-tiled: coalesced f32 reads along W, coalesced bf16 writes along C.
// ---------------------------------------------------------------------------
__global__ __launch_bounds__(256)
void transpose_kernel(const float* __restrict__ in, unsigned short* __restrict__ inT) {
    __shared__ unsigned short lds[32][33];
    const int tx = threadIdx.x;          // 32
    const int ty = threadIdx.y;          // 8
    const int tw = blockIdx.x & 1;       // w tile (2 x 32)
    const int tc = blockIdx.x >> 1;      // c tile (16 x 32)
    const int hh = blockIdx.y;           // 64
    const int b  = blockIdx.z;           // 8
    const int w0 = tw * 32, c0 = tc * 32;

    for (int j = 0; j < 4; ++j) {
        int ci = ty + 8 * j;
        lds[ci][tx] = f32_to_bf16(
            in[(((size_t)b * IN_CH + c0 + ci) * HW + hh) * HW + w0 + tx]);
    }
    __syncthreads();
    for (int j = 0; j < 4; ++j) {
        int wi = ty + 8 * j;
        inT[(((size_t)b * HP + hh + 1) * HP + (w0 + wi) + 1) * IN_CH + c0 + tx]
            = lds[tx][wi];
    }
}

// ---------------------------------------------------------------------------
// K1: style[b][i] = (w[b] . affine_weight[i]) / sqrt(512) + bias[i]
// ---------------------------------------------------------------------------
__global__ __launch_bounds__(256)
void style_kernel(const float* __restrict__ w, const float* __restrict__ aw,
                  const float* __restrict__ bias, float* __restrict__ style) {
    int gid = blockIdx.x * 256 + threadIdx.x;     // 0 .. 8*512-1
    int b = gid >> 9, i = gid & 511;
    const float* wr = w + b * 512;
    const float* ar = aw + (size_t)i * 512;
    float acc = 0.f;
    for (int l = 0; l < 512; ++l) acc += wr[l] * ar[l];
    style[gid] = acc * 0.04419417382415922f + bias[i];   // 1/sqrt(512)
}

// ---------------------------------------------------------------------------
// K2: per-(b,o) modulate + demodulate, pack bf16 as wmod[b][tap][oc][ic]
// ---------------------------------------------------------------------------
__global__ __launch_bounds__(256)
void modulate_kernel(const float* __restrict__ cw,     // (256,512,3,3)
                     const float* __restrict__ style,  // (8,512)
                     unsigned short* __restrict__ wmod) // (8,9,256,512) bf16
{
    __shared__ float red[256];
    const int o = blockIdx.x & 255;
    const int b = blockIdx.x >> 8;
    const int t = threadIdx.x;
    const float scale = 0.014731391274719740f;          // 1/sqrt(512*9)

    float vals[18];
    float ss = 0.f;
    for (int j = 0; j < 2; ++j) {
        int i = t + j * 256;
        float st = style[b * 512 + i];
        const float* base = cw + ((size_t)o * 512 + i) * 9;
        for (int tap = 0; tap < 9; ++tap) {
            float v = scale * base[tap] * st;
            vals[j * 9 + tap] = v;
            ss += v * v;
        }
    }
    red[t] = ss;
    __syncthreads();
    for (int off = 128; off > 0; off >>= 1) {
        if (t < off) red[t] += red[t + off];
        __syncthreads();
    }
    float demod = rsqrtf(red[0] + 1e-8f);

    for (int j = 0; j < 2; ++j) {
        int i = t + j * 256;
        for (int tap = 0; tap < 9; ++tap) {
            float v = vals[j * 9 + tap] * demod;
            wmod[(((size_t)b * 9 + tap) * OUT_CH + o) * IN_CH + i] = f32_to_bf16(v);
        }
    }
}

// ---------------------------------------------------------------------------
// K3: parity-decomposed stride-2 conv-transpose as implicit GEMM via WMMA.
//   ct[2a+py][2b+px] = sum_{kh%2==py, kw%2==px} in[a-(kh>>1)][b-(kw>>1)] * W[kh][kw]
// Block: 128 threads = 4 waves; wave w -> 16 pixels x 64 outch (4 accumulators).
// A frags come straight from channels-last zero-haloed global memory (2x b128),
// B frags from packed bf16 weights (2x b128 each). No LDS / barriers in hot loop.
// ---------------------------------------------------------------------------
__global__ __launch_bounds__(128)
void convt_wmma_kernel(const unsigned short* __restrict__ inT,  // (8,66,66,512) bf16
                       const unsigned short* __restrict__ wmod, // (8,9,256,512) bf16
                       float* __restrict__ ct)                  // (8,256,129,129)
{
    const int tile = blockIdx.x;
    const int pc   = blockIdx.y;          // parity class: py*2+px
    const int b    = blockIdx.z;
    const int py = pc >> 1, px = pc & 1;
    const int Hc = 65 - py, Wc = 65 - px;
    const int npix = Hc * Wc;
    if (tile * 16 >= npix) return;        // block-uniform

    const int lane = threadIdx.x & 31;
    const int wave = threadIdx.x >> 5;
    const int m    = lane & 15;
    const int g    = tile * 16 + m;
    const int gc   = (g < npix) ? g : (npix - 1);   // clamp: keep EXEC full, mask stores
    const int aRow = gc / Wc;
    const int aCol = gc - aRow * Wc;
    // A layout (16-bit 16x32): lanes 0-15 hold K{0..7,16..23}, lanes 16-31 K{8..15,24..31}
    const int koffA = (lane & 16) ? 8 : 0;
    // B layout: same as A, lanes 0-15 hold K{0..7,16..23}, lanes 16-31 K{8..15,24..31}
    const int koffB = (lane & 16) ? 8 : 0;
    const int nBase = wave * 64 + m;      // first out-channel column of this wave

    v8f acc[4] = {};

    const int nkh = py ? 1 : 2;
    const int nkw = px ? 1 : 2;
    for (int ih = 0; ih < nkh; ++ih) {
        const int kh = py ? 1 : ih * 2;
        const int hp = aRow - (kh >> 1) + 1;            // padded coords, always in-range
        for (int iw = 0; iw < nkw; ++iw) {
            const int kw = px ? 1 : iw * 2;
            const int wp = aCol - (kw >> 1) + 1;
            const int tap = kh * 3 + kw;
            const unsigned short* aBase =
                inT + (((size_t)b * HP + hp) * HP + wp) * IN_CH + koffA;
            const unsigned short* bBase =
                wmod + (((size_t)b * 9 + tap) * OUT_CH + nBase) * IN_CH + koffB;

            #pragma unroll 2
            for (int c = 0; c < IN_CH; c += 32) {
                v16bf aF;
                {   const uint4* pa = (const uint4*)(aBase + c);
                    ((uint4*)&aF)[0] = pa[0];   // K c+koffA .. +7
                    ((uint4*)&aF)[1] = pa[2];   // K c+koffA+16 .. +23
                }
                #pragma unroll
                for (int j = 0; j < 4; ++j) {
                    v16bf bF;
                    const uint4* pb = (const uint4*)(bBase + (size_t)j * 16 * IN_CH + c);
                    ((uint4*)&bF)[0] = pb[0];
                    ((uint4*)&bF)[1] = pb[2];
                    acc[j] = __builtin_amdgcn_wmma_f32_16x16x32_bf16(
                                 false, aF, false, bF, (short)0, acc[j], false, false);
                }
            }
        }
    }

    // D layout: lane -> N = lane&15 ; VGPR r -> M = r + 8*(lane>=16)
    const int mD0 = (lane & 16) ? 8 : 0;
    for (int j = 0; j < 4; ++j) {
        const int n = nBase + j * 16;
        for (int r = 0; r < 8; ++r) {
            int gg = tile * 16 + mD0 + r;
            if (gg < npix) {
                int row = gg / Wc, col = gg - row * Wc;
                int y = 2 * row + py, x = 2 * col + px;
                ct[(((size_t)b * OUT_CH + n) * CT + y) * CT + x] = acc[j][r];
            }
        }
    }
}

// ---------------------------------------------------------------------------
// K4: depthwise 4x4 blur (separable [1,3,3,1]^2 / 16), pad (1,1) -> (8,256,128,128)
// ---------------------------------------------------------------------------
__global__ __launch_bounds__(256)
void blur_kernel(const float* __restrict__ ct, float* __restrict__ out) {
    size_t idx = (size_t)blockIdx.x * 256 + threadIdx.x;
    const size_t total = (size_t)B_ * OUT_CH * OUTHW * OUTHW;
    if (idx >= total) return;
    int x  = idx & 127;
    int y  = (idx >> 7) & 127;
    size_t bo = idx >> 14;
    const float* base = ct + bo * (CT * CT);
    const float kk[4] = {1.f, 3.f, 3.f, 1.f};
    float s = 0.f;
    for (int dy = 0; dy < 4; ++dy) {
        int cy = y - 1 + dy;
        if (cy < 0 || cy >= CT) continue;
        float rs = 0.f;
        for (int dx = 0; dx < 4; ++dx) {
            int cx = x - 1 + dx;
            if (cx < 0 || cx >= CT) continue;
            rs += kk[dx] * base[cy * CT + cx];
        }
        s += kk[dy] * rs;
    }
    out[idx] = s * (1.0f / 16.0f);
}

// ---------------------------------------------------------------------------
extern "C" void kernel_launch(void* const* d_in, const int* in_sizes, int n_in,
                              void* d_out, int out_size, void* d_ws, size_t ws_size,
                              hipStream_t stream) {
    const float* in  = (const float*)d_in[0];   // (8,512,64,64)
    const float* w   = (const float*)d_in[1];   // (8,512)
    const float* aw  = (const float*)d_in[2];   // (512,512)
    const float* ab  = (const float*)d_in[3];   // (512)
    const float* cw  = (const float*)d_in[4];   // (1,256,512,3,3)
    float* out = (float*)d_out;

    char* ws = (char*)d_ws;
    float*          style = (float*)ws;                         // 16 KB
    unsigned short* wmod  = (unsigned short*)(ws + 16384);      // 18,874,368 B
    unsigned short* inT   = (unsigned short*)(ws + 18890752);   // 35,684,352 B
    float*          ct    = (float*)(ws + 54575104);            // 136,323,072 B

    // zero padded input buffer (halo), then transpose to channels-last bf16
    const unsigned int n16 = (unsigned int)((size_t)B_ * HP * HP * IN_CH * 2 / 16);
    zero_kernel<<<(n16 + 255) / 256, 256, 0, stream>>>((uint4*)inT, n16);
    {
        dim3 gt(32, 64, B_);     // (2 w-tiles * 16 c-tiles, 64 rows, 8 samples)
        dim3 bt(32, 8, 1);
        transpose_kernel<<<gt, bt, 0, stream>>>(in, inT);
    }

    style_kernel<<<16, 256, 0, stream>>>(w, aw, ab, style);
    modulate_kernel<<<B_ * OUT_CH, 256, 0, stream>>>(cw, style, wmod);

    dim3 g3(265 /* ceil(65*65/16) */, 4 /* parity */, B_);
    convt_wmma_kernel<<<g3, 128, 0, stream>>>(inT, wmod, ct);

    size_t total = (size_t)B_ * OUT_CH * OUTHW * OUTHW;
    blur_kernel<<<(unsigned)((total + 255) / 256), 256, 0, stream>>>(ct, out);
}
